// Ec2vaeDecoder_26680336843280
// MI455X (gfx1250) — matrix operand order repaired and checked by reading, not compiled, not run on hardware
//
#include <hip/hip_runtime.h>
#include <hip/hip_bf16.h>

// ---------------------------------------------------------------------------
// EC2VAE decoder on MI455X (gfx1250).
//   - weights converted once per launch to bf16 (~108 MB -> resident in 192MB L2)
//   - all GEMMs via V_WMMA_F32_16X16X32_BF16, wave32, M=64 x N=16 per wave
//   - strict sequential scan over T=32 handled by stream-ordered launches
// ---------------------------------------------------------------------------

typedef __bf16 bf16_t;
typedef __attribute__((ext_vector_type(16))) __bf16 v16bf;
typedef __attribute__((ext_vector_type(8)))  __bf16 v8bf;
typedef __attribute__((ext_vector_type(8)))  float  v8f;

#define B_    64
#define H_    2048
#define G3_   6144          // 3*H
#define T_    32
#define MEL_  130
#define RHY_  3
#define CH_   12
#define ZP_   128
#define ZR_   128
#define KRHY  160           // pad(RHY+ZR=131 -> mult of 32)
#define KFIN  288           // pad(MEL+RHY+ZP+CH=273 -> mult of 32)
#define NPRL  16            // padded rhythm logit width
#define NPML  144           // padded melody logit width

__device__ __forceinline__ bf16_t f2bf(float f) {
  unsigned u = __builtin_bit_cast(unsigned, f);
  unsigned r = u + 0x7FFFu + ((u >> 16) & 1u);   // round-to-nearest-even
  unsigned short h = (unsigned short)(r >> 16);
  return __builtin_bit_cast(bf16_t, h);
}

// ---------------------------------------------------------------------------
// GEMM: C[64, Np] = A[64, Kp](bf16, row-major) * W[Np, Kp]^T (bf16, row-major)
//                   + bias[Nreal]
// One wave computes M=64 (4 tiles) x N=16 over the whole K.  Block = 4 waves.
// A/B fragments both use contiguous k = 16*g + s (A 16x32, B 32x16); C is the 16x16 f32 tile.
// ---------------------------------------------------------------------------
__global__ __launch_bounds__(128)
void gemm64_wmma(const bf16_t* __restrict__ A, const bf16_t* __restrict__ W,
                 const float* __restrict__ bias, float* __restrict__ C,
                 int Kp, int Np, int Nreal) {
  const int wave = threadIdx.x >> 5;
  const int nb   = blockIdx.x * 64 + wave * 16;
  if (nb >= Np) return;                       // whole-wave exit keeps EXEC all-1s
  const int lane = threadIdx.x & 31;
  const int lr   = lane & 15;
  const int g    = lane >> 4;

  v8f acc0 = {}, acc1 = {}, acc2 = {}, acc3 = {};

  // B fragment: lane holds col n = nb+lr, K = kb + 16*g .. +16 (contiguous)
  const bf16_t* wp = W + (size_t)(nb + lr) * Kp + 16 * g;
  // A fragment: lane holds row m, K = kb + 16*g .. +16 (contiguous)
  const bf16_t* ap = A + (size_t)lr * Kp + 16 * g;
  const size_t tstep = (size_t)16 * Kp;       // next M-tile (16 rows)

  for (int kb = 0; kb < Kp; kb += 32) {
    v16bf bfrag = *reinterpret_cast<const v16bf*>(wp + kb);

    v8bf a0l = *reinterpret_cast<const v8bf*>(ap + kb);
    v8bf a0h = *reinterpret_cast<const v8bf*>(ap + kb + 8);
    v8bf a1l = *reinterpret_cast<const v8bf*>(ap + tstep + kb);
    v8bf a1h = *reinterpret_cast<const v8bf*>(ap + tstep + kb + 8);
    v8bf a2l = *reinterpret_cast<const v8bf*>(ap + 2 * tstep + kb);
    v8bf a2h = *reinterpret_cast<const v8bf*>(ap + 2 * tstep + kb + 8);
    v8bf a3l = *reinterpret_cast<const v8bf*>(ap + 3 * tstep + kb);
    v8bf a3h = *reinterpret_cast<const v8bf*>(ap + 3 * tstep + kb + 8);

    v16bf a0 = __builtin_shufflevector(a0l, a0h, 0,1,2,3,4,5,6,7,8,9,10,11,12,13,14,15);
    v16bf a1 = __builtin_shufflevector(a1l, a1h, 0,1,2,3,4,5,6,7,8,9,10,11,12,13,14,15);
    v16bf a2 = __builtin_shufflevector(a2l, a2h, 0,1,2,3,4,5,6,7,8,9,10,11,12,13,14,15);
    v16bf a3 = __builtin_shufflevector(a3l, a3h, 0,1,2,3,4,5,6,7,8,9,10,11,12,13,14,15);

    acc0 = __builtin_amdgcn_wmma_f32_16x16x32_bf16(false, a0, false, bfrag, (short)0, acc0, false, false);
    acc1 = __builtin_amdgcn_wmma_f32_16x16x32_bf16(false, a1, false, bfrag, (short)0, acc1, false, false);
    acc2 = __builtin_amdgcn_wmma_f32_16x16x32_bf16(false, a2, false, bfrag, (short)0, acc2, false, false);
    acc3 = __builtin_amdgcn_wmma_f32_16x16x32_bf16(false, a3, false, bfrag, (short)0, acc3, false, false);
  }

  const int n = nb + lr;
  const float bv = (bias != nullptr && n < Nreal) ? bias[n] : 0.0f;
#pragma unroll
  for (int v = 0; v < 8; ++v) {
    const int m = v + 8 * g;                 // C layout: VGPR v, lanes<16 -> M=v, else M=v+8
    C[(size_t)m        * Np + n] = acc0[v] + bv;
    C[(size_t)(m + 16) * Np + n] = acc1[v] + bv;
    C[(size_t)(m + 32) * Np + n] = acc2[v] + bv;
    C[(size_t)(m + 48) * Np + n] = acc3[v] + bv;
  }
}

// ---------------------------------------------------------------------------
// GRU gate math: gi/gh are [64, 6144] (gate order r,z,n), hprev/hout [64,2048]
// ---------------------------------------------------------------------------
__global__ __launch_bounds__(256)
void gru_gates(const float* __restrict__ gi, const float* __restrict__ gh,
               const float* __restrict__ hprev, float* __restrict__ hout,
               bf16_t* __restrict__ hout_bf) {
  int idx = blockIdx.x * 256 + threadIdx.x;
  if (idx >= B_ * H_) return;
  int b = idx / H_, j = idx - b * H_;
  size_t base = (size_t)b * G3_ + j;
  float ir = gi[base], iz = gi[base + H_], in_ = gi[base + 2 * H_];
  float hr = gh[base], hz = gh[base + H_], hn  = gh[base + 2 * H_];
  float r = 1.0f / (1.0f + expf(-(ir + hr)));
  float u = 1.0f / (1.0f + expf(-(iz + hz)));
  float n = tanhf(in_ + r * hn);
  float h = (1.0f - u) * n + u * hprev[idx];
  hout[idx]    = h;
  hout_bf[idx] = f2bf(h);
}

__global__ __launch_bounds__(256)
void tanh_init(const float* __restrict__ g, float* __restrict__ h,
               bf16_t* __restrict__ hb) {
  int idx = blockIdx.x * 256 + threadIdx.x;
  if (idx >= B_ * H_) return;
  float v = tanhf(g[idx]);
  h[idx]  = v;
  hb[idx] = f2bf(v);
}

// f32 [srcR, srcC] -> bf16 [*, dstC] zero-padded
__global__ __launch_bounds__(256)
void cvt_pad(const float* __restrict__ src, bf16_t* __restrict__ dst,
             int srcR, int srcC, int dstC, int total) {
  int idx = blockIdx.x * 256 + threadIdx.x;
  if (idx >= total) return;
  int r = idx / dstC, c = idx - r * dstC;
  float v = (r < srcR && c < srcC) ? src[(size_t)r * srcC + c] : 0.0f;
  dst[idx] = f2bf(v);
}

// place f32 [64, C] into bf16 buffer rows of given stride at column offset
__global__ __launch_bounds__(256)
void cvt_place(const float* __restrict__ src, bf16_t* __restrict__ dst,
               int C, int stride, int off) {
  int idx = blockIdx.x * 256 + threadIdx.x;
  if (idx >= B_ * C) return;
  int b = idx / C, c = idx - b * C;
  dst[(size_t)b * stride + off + c] = f2bf(src[idx]);
}

__global__ __launch_bounds__(256)
void zero_bf(bf16_t* __restrict__ p, int total) {
  int idx = blockIdx.x * 256 + threadIdx.x;
  if (idx < total) p[idx] = f2bf(0.0f);
}

__global__ void set_col(bf16_t* __restrict__ x, int stride, int col) {
  int b = threadIdx.x;
  if (b < B_) x[(size_t)b * stride + col] = f2bf(1.0f);
}

// rhythm head: log_softmax over 3, store rhy_lo[t], write one-hot feedback
__global__ void rhy_head(const float* __restrict__ lo, float* __restrict__ rlo,
                         bf16_t* __restrict__ xrhy, int t) {
  int b = threadIdx.x;
  if (b >= B_) return;
  float v0 = lo[b * NPRL + 0], v1 = lo[b * NPRL + 1], v2 = lo[b * NPRL + 2];
  float m = fmaxf(v0, fmaxf(v1, v2));
  float lse = m + logf(expf(v0 - m) + expf(v1 - m) + expf(v2 - m));
  int am = 0; float best = v0;
  if (v1 > best) { best = v1; am = 1; }
  if (v2 > best) { best = v2; am = 2; }
  float* o = rlo + (size_t)t * B_ * RHY_ + b * RHY_;
  o[0] = v0 - lse; o[1] = v1 - lse; o[2] = v2 - lse;
  bf16_t* x = xrhy + (size_t)b * KRHY;
  x[0] = f2bf(am == 0 ? 1.0f : 0.0f);
  x[1] = f2bf(am == 1 ? 1.0f : 0.0f);
  x[2] = f2bf(am == 2 ? 1.0f : 0.0f);
}

// melody head: log_softmax over 130 -> d_out[b][t][:], one-hot feedback
__global__ void mel_head(const float* __restrict__ lo, float* __restrict__ out,
                         bf16_t* __restrict__ xfin, int t) {
  int b = threadIdx.x;
  if (b >= B_) return;
  const float* row = lo + (size_t)b * NPML;
  float m = row[0];
  for (int j = 1; j < MEL_; ++j) m = fmaxf(m, row[j]);
  float s = 0.0f;
  for (int j = 0; j < MEL_; ++j) s += expf(row[j] - m);
  float lse = m + logf(s);
  int am = 0; float best = row[0];
  for (int j = 1; j < MEL_; ++j)
    if (row[j] > best) { best = row[j]; am = j; }
  float* o = out + (size_t)b * T_ * MEL_ + (size_t)t * MEL_;
  bf16_t* x = xfin + (size_t)b * KFIN;
  for (int j = 0; j < MEL_; ++j) {
    o[j] = row[j] - lse;
    x[j] = f2bf(j == am ? 1.0f : 0.0f);
  }
}

// per-step final-decoder input update: r_t (3) and c_t (12)
__global__ __launch_bounds__(256)
void fin_xupd(const float* __restrict__ rlo, const float* __restrict__ cond,
              bf16_t* __restrict__ xfin, int t) {
  int idx = blockIdx.x * 256 + threadIdx.x;
  if (idx >= B_ * 16) return;
  int b = idx >> 4, k = idx & 15;
  if (k < RHY_) {
    xfin[(size_t)b * KFIN + MEL_ + k] =
        f2bf(rlo[(size_t)t * B_ * RHY_ + b * RHY_ + k]);
  } else if (k < RHY_ + CH_) {
    int c = k - RHY_;
    xfin[(size_t)b * KFIN + (MEL_ + RHY_ + ZP_) + c] =
        f2bf(cond[(size_t)b * T_ * CH_ + (size_t)t * CH_ + c]);
  }
}

// ---------------------------------------------------------------------------
extern "C" void kernel_launch(void* const* d_in, const int* in_sizes, int n_in,
                              void* d_out, int out_size, void* d_ws, size_t ws_size,
                              hipStream_t stream) {
  (void)in_sizes; (void)n_in; (void)out_size; (void)ws_size;

  const float* z1    = (const float*)d_in[0];
  const float* z2    = (const float*)d_in[1];
  const float* cond  = (const float*)d_in[2];
  const float* w_ih0 = (const float*)d_in[3];
  const float* w_hh0 = (const float*)d_in[4];
  const float* b_ih0 = (const float*)d_in[5];
  const float* b_hh0 = (const float*)d_in[6];
  const float* w_ih1 = (const float*)d_in[7];
  const float* w_hh1 = (const float*)d_in[8];
  const float* b_ih1 = (const float*)d_in[9];
  const float* b_hh1 = (const float*)d_in[10];
  const float* w_ih2 = (const float*)d_in[11];
  const float* w_hh2 = (const float*)d_in[12];
  const float* b_ih2 = (const float*)d_in[13];
  const float* b_hh2 = (const float*)d_in[14];
  const float* wi0   = (const float*)d_in[15];
  const float* bi0   = (const float*)d_in[16];
  const float* wo0   = (const float*)d_in[17];
  const float* bo0   = (const float*)d_in[18];
  const float* wi1   = (const float*)d_in[19];
  const float* bi1   = (const float*)d_in[20];
  const float* wo1   = (const float*)d_in[21];
  const float* bo1   = (const float*)d_in[22];
  float* out = (float*)d_out;

  // ---- carve workspace (~113 MB; bf16 weights stay L2-resident) ----
  char* ws = (char*)d_ws;
  size_t off = 0;
  auto carve = [&](size_t bytes) -> char* {
    char* p = ws + off;
    off = (off + bytes + 255) & ~(size_t)255;
    return p;
  };
  bf16_t* wih0b = (bf16_t*)carve((size_t)G3_ * KRHY * 2);
  bf16_t* whh0b = (bf16_t*)carve((size_t)G3_ * H_ * 2);
  bf16_t* wih1b = (bf16_t*)carve((size_t)G3_ * KFIN * 2);
  bf16_t* whh1b = (bf16_t*)carve((size_t)G3_ * H_ * 2);
  bf16_t* wih2b = (bf16_t*)carve((size_t)G3_ * H_ * 2);
  bf16_t* whh2b = (bf16_t*)carve((size_t)G3_ * H_ * 2);
  bf16_t* wi0b  = (bf16_t*)carve((size_t)H_ * ZR_ * 2);
  bf16_t* wi1b  = (bf16_t*)carve((size_t)H_ * ZP_ * 2);
  bf16_t* wo0b  = (bf16_t*)carve((size_t)NPRL * H_ * 2);
  bf16_t* wo1b  = (bf16_t*)carve((size_t)NPML * H_ * 2);
  bf16_t* xrhy  = (bf16_t*)carve((size_t)B_ * KRHY * 2);
  bf16_t* xfin  = (bf16_t*)carve((size_t)B_ * KFIN * 2);
  bf16_t* z1b   = (bf16_t*)carve((size_t)B_ * ZP_ * 2);
  bf16_t* z2b   = (bf16_t*)carve((size_t)B_ * ZR_ * 2);
  float*  h0f   = (float*)carve((size_t)B_ * H_ * 4);
  bf16_t* h0b   = (bf16_t*)carve((size_t)B_ * H_ * 2);
  float*  haf   = (float*)carve((size_t)B_ * H_ * 4);
  bf16_t* hab   = (bf16_t*)carve((size_t)B_ * H_ * 2);
  float*  hbf   = (float*)carve((size_t)B_ * H_ * 4);
  bf16_t* hbb   = (bf16_t*)carve((size_t)B_ * H_ * 2);
  float*  gi    = (float*)carve((size_t)B_ * G3_ * 4);
  float*  gh    = (float*)carve((size_t)B_ * G3_ * 4);
  float*  lo    = (float*)carve((size_t)B_ * NPML * 4);
  float*  rlo   = (float*)carve((size_t)T_ * B_ * RHY_ * 4);

  auto cvtW = [&](const float* s, bf16_t* d, int R, int C, int Cp, int Rp) {
    int tot = Rp * Cp;
    cvt_pad<<<dim3((tot + 255) / 256), dim3(256), 0, stream>>>(s, d, R, C, Cp, tot);
  };
  auto gemm = [&](const bf16_t* A, const bf16_t* W, const float* bias, float* C,
                  int Kp, int Np, int Nreal) {
    gemm64_wmma<<<dim3((Np + 63) / 64), dim3(128), 0, stream>>>(A, W, bias, C, Kp, Np, Nreal);
  };
  const int gg = (B_ * H_ + 255) / 256;  // 512 blocks for elementwise [64,2048]

  // ---- one-time (per launch) weight conversion to bf16 ----
  cvtW(w_ih0, wih0b, G3_, ZR_ + RHY_, KRHY, G3_);
  cvtW(w_hh0, whh0b, G3_, H_, H_, G3_);
  cvtW(w_ih1, wih1b, G3_, MEL_ + RHY_ + ZP_ + CH_, KFIN, G3_);
  cvtW(w_hh1, whh1b, G3_, H_, H_, G3_);
  cvtW(w_ih2, wih2b, G3_, H_, H_, G3_);
  cvtW(w_hh2, whh2b, G3_, H_, H_, G3_);
  cvtW(wi0, wi0b, H_, ZR_, ZR_, H_);
  cvtW(wi1, wi1b, H_, ZP_, ZP_, H_);
  cvtW(wo0, wo0b, RHY_, H_, H_, NPRL);
  cvtW(wo1, wo1b, MEL_, H_, H_, NPML);

  // ---- input buffers: [onehot | z2 | pad] and [onehot | r | z1 | c | pad] ----
  zero_bf<<<dim3((B_ * KRHY + 255) / 256), dim3(256), 0, stream>>>(xrhy, B_ * KRHY);
  zero_bf<<<dim3((B_ * KFIN + 255) / 256), dim3(256), 0, stream>>>(xfin, B_ * KFIN);
  cvt_place<<<dim3((B_ * ZR_ + 255) / 256), dim3(256), 0, stream>>>(z2, xrhy, ZR_, KRHY, RHY_);
  cvt_place<<<dim3((B_ * ZP_ + 255) / 256), dim3(256), 0, stream>>>(z1, xfin, ZP_, KFIN, MEL_ + RHY_);
  cvt_place<<<dim3((B_ * ZR_ + 255) / 256), dim3(256), 0, stream>>>(z2, z2b, ZR_, ZR_, 0);
  cvt_place<<<dim3((B_ * ZP_ + 255) / 256), dim3(256), 0, stream>>>(z1, z1b, ZP_, ZP_, 0);
  set_col<<<dim3(1), dim3(64), 0, stream>>>(xrhy, KRHY, RHY_ - 1);
  set_col<<<dim3(1), dim3(64), 0, stream>>>(xfin, KFIN, MEL_ - 1);

  // ---- initial hidden states: h0 = tanh(z2 wi0^T + bi0), h1 = tanh(z1 wi1^T + bi1)
  gemm(z2b, wi0b, bi0, gi, ZR_, H_, H_);
  tanh_init<<<dim3(gg), dim3(256), 0, stream>>>(gi, h0f, h0b);
  gemm(z1b, wi1b, bi1, gi, ZP_, H_, H_);
  tanh_init<<<dim3(gg), dim3(256), 0, stream>>>(gi, haf, hab);

  // ---- rhythm decoder scan ----
  for (int t = 0; t < T_; ++t) {
    gemm(xrhy, wih0b, b_ih0, gi, KRHY, G3_, G3_);
    gemm(h0b, whh0b, b_hh0, gh, H_, G3_, G3_);
    gru_gates<<<dim3(gg), dim3(256), 0, stream>>>(gi, gh, h0f, h0f, h0b);
    gemm(h0b, wo0b, bo0, lo, H_, NPRL, RHY_);
    rhy_head<<<dim3(1), dim3(64), 0, stream>>>(lo, rlo, xrhy, t);
  }

  // ---- final (melody) decoder scan ----
  for (int t = 0; t < T_; ++t) {
    fin_xupd<<<dim3(4), dim3(256), 0, stream>>>(rlo, cond, xfin, t);
    gemm(xfin, wih1b, b_ih1, gi, KFIN, G3_, G3_);
    gemm(hab, whh1b, b_hh1, gh, H_, G3_, G3_);
    gru_gates<<<dim3(gg), dim3(256), 0, stream>>>(gi, gh, haf, haf, hab);

    const bf16_t* hbin_b = (t == 0) ? hab : hbb;   // jnp.where(i==0, ha, hb)
    const float*  hbin_f = (t == 0) ? haf : hbf;
    gemm(hab, wih2b, b_ih2, gi, H_, G3_, G3_);
    gemm(hbin_b, whh2b, b_hh2, gh, H_, G3_, G3_);
    gru_gates<<<dim3(gg), dim3(256), 0, stream>>>(gi, gh, hbin_f, hbf, hbb);

    gemm(hbb, wo1b, bo1, lo, H_, NPML, MEL_);
    mel_head<<<dim3(1), dim3(64), 0, stream>>>(lo, out, xfin, t);
  }
}
